// DeformableAttention_36764920053898
// MI455X (gfx1250) — matrix operand order repaired and checked by reading, not compiled, not run on hardware
//
#include <hip/hip_runtime.h>
#include <hip/hip_bf16.h>

typedef unsigned short u16;
typedef __attribute__((ext_vector_type(16))) __bf16 v16bf;
typedef __attribute__((ext_vector_type(8)))  float  v8f;

#define B_  8
#define S_  2048
#define E_  512
#define H_  8
#define P_  8
#define Dh_ 64
#define N_  (B_*S_)

#define USE_ASYNC_GATHER 1

__device__ __forceinline__ u16 f2b(float f) {
  unsigned u = __float_as_uint(f);
  unsigned r = (u + 0x7FFFu + ((u >> 16) & 1u)) >> 16;  // round-to-nearest-even
  return (u16)r;
}
__device__ __forceinline__ float b2f(u16 s) {
  return __uint_as_float(((unsigned)s) << 16);
}

union BfVec { uint4 u[2]; v16bf v; };

__global__ void cvt_f32_bf16(const float* __restrict__ src, u16* __restrict__ dst, int n) {
  for (int i = blockIdx.x * blockDim.x + threadIdx.x; i < n; i += gridDim.x * blockDim.x)
    dst[i] = f2b(src[i]);
}

__device__ __forceinline__ void storeOut(float* p, float v) { *p = v; }
__device__ __forceinline__ void storeOut(u16*   p, float v) { *p = f2b(v); }

// Y[N x C] = X[N x K](bf16) * W[C x K](bf16)^T + bias(f32)
// One wave -> 32 rows x (NT*16) cols: 2 A-tiles x NT B-tiles, 2*NT WMMAs/K-step.
template <typename OutT, int NT>
__global__ __launch_bounds__(128) void gemm_xwT(
    const u16* __restrict__ X, const u16* __restrict__ W,
    const float* __restrict__ bias, OutT* __restrict__ Y,
    int K, int C) {
  const int lane = threadIdx.x & 31;
  const int wave = threadIdx.x >> 5;
  const int m  = lane & 15;   // A row / B col / D col for this lane
  const int hk = lane >> 4;   // K-half selector (per ISA layout tables)
  const int rowBase = blockIdx.x * 32;
  const int colBase = (blockIdx.y * 4 + wave) * (NT * 16);

  const u16* xrow0 = X + (size_t)(rowBase + m) * K;
  const u16* xrow1 = X + (size_t)(rowBase + 16 + m) * K;

  v8f acc[2][NT];
#pragma unroll
  for (int t = 0; t < NT; ++t) {
    float bv = bias[colBase + t * 16 + m];
#pragma unroll
    for (int i = 0; i < 8; ++i) { acc[0][t][i] = bv; acc[1][t][i] = bv; }
  }

  for (int k0 = 0; k0 < K; k0 += 32) {
    BfVec a0, a1;
    a0.u[0] = *(const uint4*)(xrow0 + k0 + hk * 8);
    a0.u[1] = *(const uint4*)(xrow0 + k0 + 16 + hk * 8);
    a1.u[0] = *(const uint4*)(xrow1 + k0 + hk * 8);
    a1.u[1] = *(const uint4*)(xrow1 + k0 + 16 + hk * 8);
#pragma unroll
    for (int t = 0; t < NT; ++t) {
      const u16* wrow = W + (size_t)(colBase + t * 16 + m) * K;
      BfVec b;
      b.u[0] = *(const uint4*)(wrow + k0 + hk * 8);
      b.u[1] = *(const uint4*)(wrow + k0 + 16 + hk * 8);
      acc[0][t] = __builtin_amdgcn_wmma_f32_16x16x32_bf16(
          false, a0.v, false, b.v, (short)0, acc[0][t], false, false);
      acc[1][t] = __builtin_amdgcn_wmma_f32_16x16x32_bf16(
          false, a1.v, false, b.v, (short)0, acc[1][t], false, false);
    }
  }

#pragma unroll
  for (int mt = 0; mt < 2; ++mt) {
#pragma unroll
    for (int t = 0; t < NT; ++t) {
#pragma unroll
      for (int j = 0; j < 8; ++j) {
        int row = rowBase + mt * 16 + j + 8 * hk;
        int col = colBase + t * 16 + m;
        storeOut(Y + (size_t)row * C + col, acc[mt][t][j]);
      }
    }
  }
}

// 16-column variant for the offsets GEMM (q @ offset_w^T), fp32 output.
__global__ __launch_bounds__(128) void gemm_xwT_c16(
    const u16* __restrict__ X, const u16* __restrict__ W,
    const float* __restrict__ bias, float* __restrict__ Y, int K) {
  const int lane = threadIdx.x & 31;
  const int wave = threadIdx.x >> 5;
  const int m  = lane & 15;
  const int hk = lane >> 4;
  const int rowBase = (blockIdx.x * 4 + wave) * 16;
  const u16* xrow = X + (size_t)(rowBase + m) * K;

  v8f acc;
  float bv = bias[m];
#pragma unroll
  for (int i = 0; i < 8; ++i) acc[i] = bv;

  const u16* wrow = W + (size_t)m * K;
  for (int k0 = 0; k0 < K; k0 += 32) {
    BfVec a, b;
    a.u[0] = *(const uint4*)(xrow + k0 + hk * 8);
    a.u[1] = *(const uint4*)(xrow + k0 + 16 + hk * 8);
    b.u[0] = *(const uint4*)(wrow + k0 + hk * 8);
    b.u[1] = *(const uint4*)(wrow + k0 + 16 + hk * 8);
    acc = __builtin_amdgcn_wmma_f32_16x16x32_bf16(
        false, a.v, false, b.v, (short)0, acc, false, false);
  }
#pragma unroll
  for (int j = 0; j < 8; ++j)
    Y[(size_t)(rowBase + j + 8 * hk) * 16 + m] = acc[j];
}

// Gather one 1KB bf16 row (512 elems) into LDS, 16B per lane per issue.
__device__ __forceinline__ void gather_row(const u16* __restrict__ src,
                                           u16* ldsDst, int lane) {
#if USE_ASYNC_GATHER
  unsigned la = (unsigned)(uintptr_t)ldsDst + (unsigned)lane * 16u;
  unsigned la2 = la + 512u;
  const u16* g  = src + lane * 8;
  const u16* g2 = g + 256;
  asm volatile("global_load_async_to_lds_b128 %0, %1, off"
               :: "v"(la), "v"(g) : "memory");
  asm volatile("global_load_async_to_lds_b128 %0, %1, off"
               :: "v"(la2), "v"(g2) : "memory");
#else
  const uint4* s4 = (const uint4*)src;
  uint4* d4 = (uint4*)ldsDst;
  d4[lane]      = s4[lane];
  d4[lane + 32] = s4[lane + 32];
#endif
}

__device__ __forceinline__ void gather_fence() {
#if USE_ASYNC_GATHER
  asm volatile("s_wait_asynccnt 0x0" ::: "memory");
#endif
}

// One wave per token: gather projected K rows -> LDS, scores + softmax over P,
// then reuse LDS for projected V rows, weighted sum -> ctx (bf16).
__global__ __launch_bounds__(128) void deform_attn(
    const u16* __restrict__ qpb, const u16* __restrict__ kpb,
    const u16* __restrict__ vpb, const float* __restrict__ offs,
    u16* __restrict__ ctxb) {
  __shared__ __align__(16) u16 kvbuf[4][P_][E_];   // 32 KB
  __shared__ float wbuf[4][H_ * P_];               // 1 KB
  const int lane = threadIdx.x & 31;
  const int wave = threadIdx.x >> 5;
  const int token = blockIdx.x * 4 + wave;
  const int bi = token / S_;
  const int si = token - bi * S_;

  int idxp[P_];
#pragma unroll
  for (int p = 0; p < P_; ++p) {
    float off = offs[(size_t)token * (2 * P_) + 2 * p];
    int id = (int)truncf((float)si + off);
    id = id < 0 ? 0 : (id > S_ - 1 ? S_ - 1 : id);
    idxp[p] = id;
  }

#pragma unroll
  for (int p = 0; p < P_; ++p)
    gather_row(kpb + ((size_t)(bi * S_ + idxp[p])) * E_, &kvbuf[wave][p][0], lane);
  gather_fence();
  __syncthreads();

  {
    const int h  = lane >> 2;
    const int pq = (lane & 3) * 2;
    const u16* qrow = qpb + (size_t)token * E_ + h * Dh_;
    const u16* k0r = &kvbuf[wave][pq][h * Dh_];
    const u16* k1r = &kvbuf[wave][pq + 1][h * Dh_];
    float s0 = 0.f, s1 = 0.f;
#pragma unroll
    for (int d = 0; d < Dh_; ++d) {
      float qv = b2f(qrow[d]);
      s0 += qv * b2f(k0r[d]);
      s1 += qv * b2f(k1r[d]);
    }
    s0 *= 0.125f; s1 *= 0.125f;
    float mx = fmaxf(s0, s1);
    mx = fmaxf(mx, __shfl_xor(mx, 1, 4));
    mx = fmaxf(mx, __shfl_xor(mx, 2, 4));
    float e0 = expf(s0 - mx), e1 = expf(s1 - mx);
    float sum = e0 + e1;
    sum += __shfl_xor(sum, 1, 4);
    sum += __shfl_xor(sum, 2, 4);
    float inv = 1.0f / sum;
    wbuf[wave][h * P_ + pq]     = e0 * inv;
    wbuf[wave][h * P_ + pq + 1] = e1 * inv;
  }
  __syncthreads();

#pragma unroll
  for (int p = 0; p < P_; ++p)
    gather_row(vpb + ((size_t)(bi * S_ + idxp[p])) * E_, &kvbuf[wave][p][0], lane);
  gather_fence();
  __syncthreads();

#pragma unroll
  for (int i = 0; i < 16; ++i) {
    int e = lane * 16 + i;
    int h = e >> 6;
    float acc = 0.f;
#pragma unroll
    for (int p = 0; p < P_; ++p)
      acc += wbuf[wave][h * P_ + p] * b2f(kvbuf[wave][p][e]);
    ctxb[(size_t)token * E_ + e] = f2b(acc);
  }
}

extern "C" void kernel_launch(void* const* d_in, const int* in_sizes, int n_in,
                              void* d_out, int out_size, void* d_ws, size_t ws_size,
                              hipStream_t stream) {
  (void)in_sizes; (void)n_in; (void)out_size; (void)ws_size;
  const float* q    = (const float*)d_in[0];
  const float* k    = (const float*)d_in[1];
  const float* v    = (const float*)d_in[2];
  const float* offw = (const float*)d_in[3];
  const float* offb = (const float*)d_in[4];
  const float* inw  = (const float*)d_in[5];
  const float* inb  = (const float*)d_in[6];
  const float* outw = (const float*)d_in[7];
  const float* outb = (const float*)d_in[8];
  float* out = (float*)d_out;

  const size_t NE = (size_t)N_ * E_;
  u16* qb    = (u16*)d_ws;
  u16* kb    = qb + NE;
  u16* vb    = kb + NE;
  u16* qpb   = vb + NE;
  u16* kpb   = qpb + NE;
  u16* vpb   = kpb + NE;
  u16* ctxb  = vpb + NE;
  u16* winb  = ctxb + NE;                    // 3*E*E  (wq|wk|wv rows)
  u16* wob   = winb + (size_t)3 * E_ * E_;   // E*E
  u16* woffb = wob + (size_t)E_ * E_;        // 2P*E
  float* offs = (float*)(((uintptr_t)(woffb + 2 * P_ * E_) + 255) & ~(uintptr_t)255);

  auto cvt = [&](const float* s, u16* d, int n) {
    int blocks = (n + 1023) / 1024; if (blocks > 4096) blocks = 4096;
    cvt_f32_bf16<<<dim3(blocks), dim3(256), 0, stream>>>(s, d, n);
  };
  cvt(q, qb, (int)NE);
  cvt(k, kb, (int)NE);
  cvt(v, vb, (int)NE);
  cvt(inw, winb, 3 * E_ * E_);
  cvt(outw, wob, E_ * E_);
  cvt(offw, woffb, 2 * P_ * E_);

  dim3 gblk(128);
  dim3 ggrid(N_ / 32, E_ / 256);  // wave tile 32x64, 4 waves/block
  gemm_xwT<u16, 4><<<ggrid, gblk, 0, stream>>>(qb, winb,                       inb,          qpb, E_, E_);
  gemm_xwT<u16, 4><<<ggrid, gblk, 0, stream>>>(kb, winb + (size_t)E_ * E_,     inb + E_,     kpb, E_, E_);
  gemm_xwT<u16, 4><<<ggrid, gblk, 0, stream>>>(vb, winb + (size_t)2 * E_ * E_, inb + 2 * E_, vpb, E_, E_);
  gemm_xwT_c16<<<dim3(N_ / 64), gblk, 0, stream>>>(qb, woffb, offb, offs, E_);
  deform_attn<<<dim3(N_ / 4), gblk, 0, stream>>>(qpb, kpb, vpb, offs, ctxb);
  gemm_xwT<float, 4><<<ggrid, gblk, 0, stream>>>(ctxb, wob, outb, out, E_, E_);
}
